// Transformer_34634616275013
// MI455X (gfx1250) — matrix operand order repaired and checked by reading, not compiled, not run on hardware
//
#include <hip/hip_runtime.h>
#include <cstdint>

#define D_MODEL 512
#define N_HEAD  8
#define D_HEAD  64
#define D_FF    2048
#define N_LAYER 6
#define NB      4
#define SEQ     512
#define VOCAB   32000
#define BS_ROWS (NB*SEQ)

typedef __attribute__((ext_vector_type(16))) __bf16 v16bf;
typedef __attribute__((ext_vector_type(8)))  float  v8f;

// ---------------- fp32 -> bf16 (round to nearest even) ----------------
__device__ __forceinline__ uint16_t f2bf(float f) {
    uint32_t u = __builtin_bit_cast(uint32_t, f);
    u += 0x7FFFu + ((u >> 16) & 1u);
    return (uint16_t)(u >> 16);
}
__device__ __forceinline__ uint32_t packbf2(float lo, float hi) {
    return (uint32_t)f2bf(lo) | ((uint32_t)f2bf(hi) << 16);
}

// LDS tile rows hold 64 bf16 K-values, row stride 68 halves (=34 dwords, padded).
// A fragment (16x32, lane=M in two groups, regs=K interleaved in 8-blocks):
//   lane l: m=l&15, g=l>>4; VGPR j(0..3): K = g*8+2j,2j+1 ; VGPR j(4..7): K = 16+g*8+...
__device__ __forceinline__ v16bf load_a_frag(const uint16_t* s, int tile, int ks, int lane) {
    const int row = tile * 16 + (lane & 15);
    const int g   = lane >> 4;
    const uint32_t* base = (const uint32_t*)(s + row * 68) + ks * 16;
    union { uint32_t u[8]; v16bf v; } f;
#pragma unroll
    for (int j = 0; j < 4; ++j) { f.u[j] = base[g * 4 + j]; f.u[4 + j] = base[8 + g * 4 + j]; }
    return f.v;
}
// B fragment (32x16, lane=N in two groups, regs=K interleaved in 8-blocks); LDS holds B^T: sB[n][k].
__device__ __forceinline__ v16bf load_b_frag(const uint16_t* s, int tile, int ks, int lane) {
    const int row = tile * 16 + (lane & 15);
    const int g   = lane >> 4;
    const uint32_t* base = (const uint32_t*)(s + row * 68) + ks * 16;
    union { uint32_t u[8]; v16bf v; } f;
#pragma unroll
    for (int j = 0; j < 4; ++j) { f.u[j] = base[g * 4 + j]; f.u[4 + j] = base[8 + g * 4 + j]; }
    return f.v;
}

__device__ __forceinline__ v8f wmma_bf16(v16bf a, v16bf b, v8f c) {
    return __builtin_amdgcn_wmma_f32_16x16x32_bf16(false, a, false, b, (short)0, c, false, false);
}

// ---------------- batched GEMM: C = A @ B(^T) (+bias) (+R) (+ReLU) ----------------
// Macro tile 128(M) x 64(N); K stage = 64 (2 x 32-K WMMA substeps, 8 WMMAs/stage/wave-tile set);
// double-buffered LDS -> one barrier per stage with prefetch overlap.
// Batch z: offA = (z/inner)*sAo + (z%inner)*sAi, same for B, C; bias offset z*sBias.
// Requires: M%128==0, N%64==0, K%64==0 (true for every GEMM in this model).
template <bool TB, bool HB, bool HR, bool RELU>
__global__ __launch_bounds__(256) void gemm_wmma_kernel(
    const float* __restrict__ A, const float* __restrict__ Bm,
    const float* __restrict__ bias, const float* __restrict__ R,
    float* __restrict__ C,
    int K, int lda, int ldb, int ldc,
    long long sAo, long long sAi, long long sBo, long long sBi,
    long long sCo, long long sCi, long long sBias,
    int inner)
{
    __shared__ uint16_t sA[2][128 * 68];
    __shared__ uint16_t sB[2][64 * 68];

    const int z  = blockIdx.z;
    const int zo = z / inner, zi = z - zo * inner;
    const float* Ab = A  + zo * sAo + (long long)zi * sAi;
    const float* Bb = Bm + zo * sBo + (long long)zi * sBi;
    const long long offC = zo * sCo + (long long)zi * sCi;

    const int n0 = blockIdx.x * 64;
    const int m0 = blockIdx.y * 128;
    const int t  = threadIdx.x;
    const int lane = t & 31, w = t >> 5;
    const int wmg = w >> 1;        // 0..3 : 32-row group
    const int wng = w & 1;         // 0..1 : 32-col group

    // loader indices
    const int ar = t >> 1, ac = (t & 1) * 32;   // A: 128 rows x 64 k, 32 floats/thread
    const int br = t >> 2, bc = (t & 3) * 16;   // B: 64 rows x 64 cols, 16 floats/thread

    auto load_stage = [&](int buf, int k0) {
        {   // A tile: sA[m][k], contiguous 128B global reads per thread
            const float* src = Ab + (long long)(m0 + ar) * lda + (k0 + ac);
            uint32_t* dst = (uint32_t*)(sA[buf] + ar * 68) + (ac >> 1);
#pragma unroll
            for (int j = 0; j < 16; ++j) dst[j] = packbf2(src[2 * j], src[2 * j + 1]);
        }
        if (TB) { // B is [N,K] row-major: sB[n][k] contiguous along k
            const float* src = Bb + (long long)(n0 + br) * ldb + (k0 + bc);
            uint32_t* dst = (uint32_t*)(sB[buf] + br * 68) + (bc >> 1);
#pragma unroll
            for (int j = 0; j < 8; ++j) dst[j] = packbf2(src[2 * j], src[2 * j + 1]);
        } else {  // B is [K,N] row-major: transpose into sB[n][k]
            const float* src = Bb + (long long)(k0 + br) * ldb + (n0 + bc);
#pragma unroll
            for (int j = 0; j < 16; ++j) sB[buf][(bc + j) * 68 + br] = f2bf(src[j]);
        }
    };

    v8f acc00 = {}, acc01 = {}, acc10 = {}, acc11 = {};

    load_stage(0, 0);
    __syncthreads();
    const int nstage = K >> 6;
    for (int s = 0; s < nstage; ++s) {
        const int buf = s & 1;
        if (s + 1 < nstage) load_stage(buf ^ 1, (s + 1) << 6);   // prefetch overlaps WMMAs
#pragma unroll
        for (int ks = 0; ks < 2; ++ks) {
            v16bf a0 = load_a_frag(sA[buf], wmg * 2,     ks, lane);
            v16bf a1 = load_a_frag(sA[buf], wmg * 2 + 1, ks, lane);
            v16bf b0 = load_b_frag(sB[buf], wng * 2,     ks, lane);
            v16bf b1 = load_b_frag(sB[buf], wng * 2 + 1, ks, lane);
            acc00 = wmma_bf16(a0, b0, acc00);
            acc01 = wmma_bf16(a0, b1, acc01);
            acc10 = wmma_bf16(a1, b0, acc10);
            acc11 = wmma_bf16(a1, b1, acc11);
        }
        __syncthreads();
    }

    // Straight-line epilogue. C/D layout: VGPR r -> M = r (lanes 0-15) or r+8 (lanes 16-31); N = lane&15.
    const int g = lane >> 4, nl = lane & 15;
    const float* biasp = HB ? (bias + (long long)z * sBias) : nullptr;
#pragma unroll
    for (int mi = 0; mi < 2; ++mi) {
#pragma unroll
        for (int ni = 0; ni < 2; ++ni) {
            v8f acc = (mi == 0) ? (ni == 0 ? acc00 : acc01) : (ni == 0 ? acc10 : acc11);
            const int nn = n0 + (wng * 2 + ni) * 16 + nl;
            const int mb = m0 + (wmg * 2 + mi) * 16 + g * 8;
            const float bv = HB ? biasp[nn] : 0.0f;
            const long long base = offC + (long long)mb * ldc + nn;
            float* cp = C + base;
            const float* rp = HR ? (R + base) : nullptr;
#pragma unroll
            for (int r = 0; r < 8; ++r) {
                float c = acc[r] + bv;
                if (HR)   c += rp[(long long)r * ldc];
                if (RELU) c = fmaxf(c, 0.0f);
                cp[(long long)r * ldc] = c;
            }
        }
    }
}

// ---------------- softmax over full row, then mask (faithful: mask AFTER softmax) ----------------
__global__ __launch_bounds__(256) void softmax_mask_kernel(
    float* __restrict__ sc, const int* __restrict__ qtok, const int* __restrict__ ktok,
    int Sq, int Sk, int Hn, int causal, float scale)
{
    const int row = blockIdx.x;
    const int z   = blockIdx.y;
    const int b   = z / Hn;
    float* p = sc + ((long long)z * Sq + row) * Sk;
    const int t = threadIdx.x;
    __shared__ float red[256];

    float mx = -3.4e38f;
    for (int c = t; c < Sk; c += 256) mx = fmaxf(mx, p[c] * scale);
    red[t] = mx; __syncthreads();
    for (int s = 128; s > 0; s >>= 1) { if (t < s) red[t] = fmaxf(red[t], red[t + s]); __syncthreads(); }
    const float rowmax = red[0]; __syncthreads();

    float sum = 0.0f;
    for (int c = t; c < Sk; c += 256) { float e = __expf(p[c] * scale - rowmax); p[c] = e; sum += e; }
    red[t] = sum; __syncthreads();
    for (int s = 128; s > 0; s >>= 1) { if (t < s) red[t] += red[t + s]; __syncthreads(); }
    const float inv = 1.0f / red[0];

    const int qnz = (qtok[(long long)b * Sq + row] != 0);
    for (int c = t; c < Sk; c += 256) {
        const int keep = qnz && (ktok[(long long)b * Sk + c] != 0) && (!causal || c <= row);
        p[c] = keep ? p[c] * inv : 1e-9f;
    }
}

// ---------------- out = LayerNorm(a + x) * gamma + beta  (D fixed = 512) ----------------
__global__ __launch_bounds__(256) void ln_res_kernel(
    const float* __restrict__ a, const float* __restrict__ x,
    const float* __restrict__ gamma, const float* __restrict__ beta,
    float* __restrict__ out)
{
    const long long row = blockIdx.x;
    const float* ap = a + row * D_MODEL;
    const float* xp = x + row * D_MODEL;
    float*       op = out + row * D_MODEL;
    const int t = threadIdx.x;
    __shared__ float red[256];

    float v0 = ap[t] + xp[t];
    float v1 = ap[t + 256] + xp[t + 256];
    red[t] = v0 + v1; __syncthreads();
    for (int s = 128; s > 0; s >>= 1) { if (t < s) red[t] += red[t + s]; __syncthreads(); }
    const float mu = red[0] * (1.0f / D_MODEL); __syncthreads();

    float d0 = v0 - mu, d1 = v1 - mu;
    red[t] = d0 * d0 + d1 * d1; __syncthreads();
    for (int s = 128; s > 0; s >>= 1) { if (t < s) red[t] += red[t + s]; __syncthreads(); }
    const float inv = rsqrtf(red[0] * (1.0f / D_MODEL) + 1e-5f);

    op[t]       = d0 * inv * gamma[t]       + beta[t];
    op[t + 256] = d1 * inv * gamma[t + 256] + beta[t + 256];
}

// ---------------- h = emb[token] + PE (base 1000, faithful to source) ----------------
__global__ __launch_bounds__(256) void embed_pe_kernel(
    const int* __restrict__ tok, const float* __restrict__ emb,
    float* __restrict__ h, long long total)
{
    const long long idx = (long long)blockIdx.x * 256 + threadIdx.x;
    if (idx >= total) return;
    const int d = (int)(idx % D_MODEL);
    const long long row = idx / D_MODEL;
    const int s = (int)(row % SEQ);
    const int tk = tok[row];
    const int j = d >> 1;
    const float dv = __expf(-(2.0f * (float)j / (float)D_MODEL) * 6.90775527898f); // ln(1000)
    const float ang = (float)s * dv;
    const float pe = (d & 1) ? __cosf(ang) : __sinf(ang);
    h[idx] = emb[(long long)tk * D_MODEL + d] + pe;
}

// ---------------- host-side orchestration ----------------
static inline void launch_gemm(hipStream_t stream, bool tb,
    const float* A, const float* Bm, const float* bias, const float* R, float* C,
    int M, int N, int K, int lda, int ldb, int ldc,
    long long sAo, long long sAi, long long sBo, long long sBi,
    long long sCo, long long sCi, long long sBias,
    int inner, int batch, int relu)
{
    dim3 grid((unsigned)(N / 64), (unsigned)(M / 128), (unsigned)batch), blk(256);
#define GEMM_ARGS A, Bm, bias, R, C, K, lda, ldb, ldc, sAo, sAi, sBo, sBi, sCo, sCi, sBias, inner
    if (tb)
        gemm_wmma_kernel<true,  false, false, false><<<grid, blk, 0, stream>>>(GEMM_ARGS);
    else if (!bias && !R)
        gemm_wmma_kernel<false, false, false, false><<<grid, blk, 0, stream>>>(GEMM_ARGS);
    else if (bias && !R && relu)
        gemm_wmma_kernel<false, true,  false, true ><<<grid, blk, 0, stream>>>(GEMM_ARGS);
    else if (bias && R)
        gemm_wmma_kernel<false, true,  true,  false><<<grid, blk, 0, stream>>>(GEMM_ARGS);
    else
        gemm_wmma_kernel<false, true,  false, false><<<grid, blk, 0, stream>>>(GEMM_ARGS);
#undef GEMM_ARGS
}

extern "C" void kernel_launch(void* const* d_in, const int* in_sizes, int n_in,
                              void* d_out, int out_size, void* d_ws, size_t ws_size,
                              hipStream_t stream)
{
    (void)in_sizes; (void)n_in; (void)out_size; (void)ws_size;

    const int*   x       = (const int*)d_in[0];
    const int*   y       = (const int*)d_in[1];
    const float* emb_src = (const float*)d_in[2];
    const float* emb_tgt = (const float*)d_in[3];
    const float* enc_qkv_w = (const float*)d_in[4];
    const float* enc_qkv_b = (const float*)d_in[5];
    const float* enc_ln_g  = (const float*)d_in[6];
    const float* enc_ln_b  = (const float*)d_in[7];
    const float* enc_ff1_w = (const float*)d_in[8];
    const float* enc_ff1_b = (const float*)d_in[9];
    const float* enc_ff2_w = (const float*)d_in[10];
    const float* enc_ff2_b = (const float*)d_in[11];
    const float* dec_self_qkv_w  = (const float*)d_in[12];
    const float* dec_self_qkv_b  = (const float*)d_in[13];
    const float* dec_cross_qkv_w = (const float*)d_in[14];
    const float* dec_cross_qkv_b = (const float*)d_in[15];
    const float* dec_ln_g  = (const float*)d_in[16];
    const float* dec_ln_b  = (const float*)d_in[17];
    const float* dec_ff1_w = (const float*)d_in[18];
    const float* dec_ff1_b = (const float*)d_in[19];
    const float* dec_ff2_w = (const float*)d_in[20];
    const float* dec_ff2_b = (const float*)d_in[21];
    const float* out_w = (const float*)d_in[22];
    const float* out_b = (const float*)d_in[23];

    const long long nBSD = (long long)BS_ROWS * D_MODEL;       // 2048*512
    float* ws   = (float*)d_ws;
    float* hs   = ws;                                          // [BS, D]
    float* ht   = hs + nBSD;                                   // [BS, D]
    float* qkv  = ht + nBSD;                                   // [3, BS, D]
    float* tmp1 = qkv + 3 * nBSD;                              // [BS, D]
    float* tmp2 = tmp1 + nBSD;                                 // [BS, D]
    float* tmp3 = tmp2 + nBSD;                                 // [BS, D]
    float* mid  = tmp3 + nBSD;                                 // [BS, DFF]
    float* scr  = mid + (long long)BS_ROWS * D_FF;             // [B, H, S, S]

    const float scale = 0.04419417382415922f;                  // 1/sqrt(512) (full D, faithful)
    const long long sHD  = (long long)SEQ * D_MODEL;           // per-batch activation stride
    const long long sHSS = (long long)N_HEAD * SEQ * SEQ;      // per-batch score stride
    const long long sSS  = (long long)SEQ * SEQ;               // per-head score stride
    const long long dd   = (long long)D_MODEL * D_MODEL;

    // embeddings + positional encoding
    {
        const int blocks = (int)((nBSD + 255) / 256);
        embed_pe_kernel<<<blocks, 256, 0, stream>>>(x, emb_src, hs, nBSD);
        embed_pe_kernel<<<blocks, 256, 0, stream>>>(y, emb_tgt, ht, nBSD);
    }

    // ---------------- encoder ----------------
    for (int l = 0; l < N_LAYER; ++l) {
        const float* wq = enc_qkv_w + (long long)l * 3 * dd;
        const float* bq = enc_qkv_b + (long long)l * 3 * D_MODEL;
        // q,k,v projections (batch over the 3 weight matrices)
        launch_gemm(stream, false, hs, wq, bq, nullptr, qkv,
                    BS_ROWS, D_MODEL, D_MODEL, D_MODEL, D_MODEL, D_MODEL,
                    0, 0, dd, 0, nBSD, 0, D_MODEL, 1, 3, 0);
        const float* q = qkv, *k = qkv + nBSD, *v = qkv + 2 * nBSD;
        // scores = Q @ K^T  (batched over b,h with head-strided views)
        launch_gemm(stream, true, q, k, nullptr, nullptr, scr,
                    SEQ, SEQ, D_HEAD, D_MODEL, D_MODEL, SEQ,
                    sHD, D_HEAD, sHD, D_HEAD, sHSS, sSS, 0, N_HEAD, NB * N_HEAD, 0);
        softmax_mask_kernel<<<dim3(SEQ, NB * N_HEAD), 256, 0, stream>>>(scr, x, x, SEQ, SEQ, N_HEAD, 0, scale);
        // ctx = P @ V
        launch_gemm(stream, false, scr, v, nullptr, nullptr, tmp1,
                    SEQ, D_HEAD, SEQ, SEQ, D_MODEL, D_MODEL,
                    sHSS, sSS, sHD, D_HEAD, sHD, D_HEAD, 0, N_HEAD, NB * N_HEAD, 0);
        // h = LN(ctx + h_src)
        ln_res_kernel<<<BS_ROWS, 256, 0, stream>>>(tmp1, hs, enc_ln_g + l * D_MODEL, enc_ln_b + l * D_MODEL, tmp2);
        // h_src = FFN(h) + h
        launch_gemm(stream, false, tmp2, enc_ff1_w + (long long)l * D_MODEL * D_FF, enc_ff1_b + (long long)l * D_FF,
                    nullptr, mid, BS_ROWS, D_FF, D_MODEL, D_MODEL, D_FF, D_FF,
                    0, 0, 0, 0, 0, 0, 0, 1, 1, 1);
        launch_gemm(stream, false, mid, enc_ff2_w + (long long)l * D_FF * D_MODEL, enc_ff2_b + (long long)l * D_MODEL,
                    tmp2, hs, BS_ROWS, D_MODEL, D_FF, D_FF, D_MODEL, D_MODEL,
                    0, 0, 0, 0, 0, 0, 0, 1, 1, 0);
    }

    // ---------------- decoder ----------------
    for (int l = 0; l < N_LAYER; ++l) {
        const float* gl = dec_ln_g + l * D_MODEL;
        const float* bl = dec_ln_b + l * D_MODEL;
        // self attention
        const float* ws_ = dec_self_qkv_w + (long long)l * 3 * dd;
        const float* bs_ = dec_self_qkv_b + (long long)l * 3 * D_MODEL;
        launch_gemm(stream, false, ht, ws_, bs_, nullptr, qkv,
                    BS_ROWS, D_MODEL, D_MODEL, D_MODEL, D_MODEL, D_MODEL,
                    0, 0, dd, 0, nBSD, 0, D_MODEL, 1, 3, 0);
        launch_gemm(stream, true, qkv, qkv + nBSD, nullptr, nullptr, scr,
                    SEQ, SEQ, D_HEAD, D_MODEL, D_MODEL, SEQ,
                    sHD, D_HEAD, sHD, D_HEAD, sHSS, sSS, 0, N_HEAD, NB * N_HEAD, 0);
        softmax_mask_kernel<<<dim3(SEQ, NB * N_HEAD), 256, 0, stream>>>(scr, y, y, SEQ, SEQ, N_HEAD, 1, scale);
        launch_gemm(stream, false, scr, qkv + 2 * nBSD, nullptr, nullptr, tmp1,
                    SEQ, D_HEAD, SEQ, SEQ, D_MODEL, D_MODEL,
                    sHSS, sSS, sHD, D_HEAD, sHD, D_HEAD, 0, N_HEAD, NB * N_HEAD, 0);
        ln_res_kernel<<<BS_ROWS, 256, 0, stream>>>(tmp1, ht, gl, bl, tmp2);   // h
        // cross attention: Q from h (tmp2), K/V from encoder output (hs)
        const float* wc = dec_cross_qkv_w + (long long)l * 3 * dd;
        const float* bc = dec_cross_qkv_b + (long long)l * 3 * D_MODEL;
        launch_gemm(stream, false, tmp2, wc, bc, nullptr, qkv,
                    BS_ROWS, D_MODEL, D_MODEL, D_MODEL, D_MODEL, D_MODEL,
                    0, 0, 0, 0, 0, 0, D_MODEL, 1, 1, 0);
        launch_gemm(stream, false, hs, wc + dd, bc + D_MODEL, nullptr, qkv + nBSD,
                    BS_ROWS, D_MODEL, D_MODEL, D_MODEL, D_MODEL, D_MODEL,
                    0, 0, dd, 0, nBSD, 0, D_MODEL, 1, 2, 0);
        launch_gemm(stream, true, qkv, qkv + nBSD, nullptr, nullptr, scr,
                    SEQ, SEQ, D_HEAD, D_MODEL, D_MODEL, SEQ,
                    sHD, D_HEAD, sHD, D_HEAD, sHSS, sSS, 0, N_HEAD, NB * N_HEAD, 0);
        softmax_mask_kernel<<<dim3(SEQ, NB * N_HEAD), 256, 0, stream>>>(scr, y, x, SEQ, SEQ, N_HEAD, 1, scale);
        launch_gemm(stream, false, scr, qkv + 2 * nBSD, nullptr, nullptr, tmp1,
                    SEQ, D_HEAD, SEQ, SEQ, D_MODEL, D_MODEL,
                    sHSS, sSS, sHD, D_HEAD, sHD, D_HEAD, 0, N_HEAD, NB * N_HEAD, 0);
        ln_res_kernel<<<BS_ROWS, 256, 0, stream>>>(tmp1, tmp2, gl, bl, tmp3); // h2 (shared LN params, faithful)
        // h_tgt = FFN(h2) + h2
        launch_gemm(stream, false, tmp3, dec_ff1_w + (long long)l * D_MODEL * D_FF, dec_ff1_b + (long long)l * D_FF,
                    nullptr, mid, BS_ROWS, D_FF, D_MODEL, D_MODEL, D_FF, D_FF,
                    0, 0, 0, 0, 0, 0, 0, 1, 1, 1);
        launch_gemm(stream, false, mid, dec_ff2_w + (long long)l * D_FF * D_MODEL, dec_ff2_b + (long long)l * D_MODEL,
                    tmp3, ht, BS_ROWS, D_MODEL, D_FF, D_FF, D_MODEL, D_MODEL,
                    0, 0, 0, 0, 0, 0, 0, 1, 1, 0);
    }

    // output projection: [2048,512] @ [512,32000] + out_b -> d_out
    launch_gemm(stream, false, ht, out_w, out_b, nullptr, (float*)d_out,
                BS_ROWS, VOCAB, D_MODEL, D_MODEL, VOCAB, VOCAB,
                0, 0, 0, 0, 0, 0, 0, 1, 1, 0);
}
